// SResTransformer_5798205849760
// MI455X (gfx1250) — matrix operand order repaired and checked by reading, not compiled, not run on hardware
//
#include <hip/hip_runtime.h>
#include <hip/hip_bf16.h>
#include <math.h>

#define BB      4
#define LL      1024
#define DMODEL  256
#define DINNER  512
#define NSTATE  16
#define DTRANK  16
#define KCONV   4
#define NLAYERS 8
#define ROWS    (BB * LL)          // 4096
#define SPDIM   (DTRANK + 2 * NSTATE)  // 48

typedef __bf16 bf16_t;
typedef __attribute__((ext_vector_type(16))) __bf16 v16bf;
typedef __attribute__((ext_vector_type(8)))  __bf16 v8bf;
typedef __attribute__((ext_vector_type(8)))  float  v8f;

// ---------------------------------------------------------------------------
// WMMA GEMM: D[M,N] = A[M,K](bf16) * Bt[N,K](bf16)^T (+ Cin[M,N] if non-null)
// Register-blocked: one wave computes a 16 x (16*JT) strip with JT independent
// 16x16 accumulators -> the A fragment is loaded once per K-step and reused JT
// times, and JT independent WMMA chains keep the XDL pipelined. JT is a
// compile-time constant so the inner loop is straight-line (loads clause,
// partial s_wait_loadcnt, no accumulator copies). The only guard is the
// scalar wave prologue check; EXEC is all-1s around every WMMA.
// Requirements: K % 32 == 0, M % 16 == 0, all JT tiles of a launched wave
// in-range (caller sizes grid/block accordingly).
// ---------------------------------------------------------------------------
template <int JT>
__global__ __launch_bounds__(256) void wmma_gemm_bf16(
    const bf16_t* __restrict__ A, const bf16_t* __restrict__ Bt,
    const float* __restrict__ Cin, float* __restrict__ D,
    int M, int N, int K)
{
  const int lane = threadIdx.x & 31;
  // Force scalar: every lane of a wave has the same threadIdx.x>>5.
  const int wave = __builtin_amdgcn_readfirstlane(threadIdx.x >> 5);
  const int half = lane >> 4;     // 0: lanes 0-15, 1: lanes 16-31
  const int mn   = lane & 15;
  const int mt   = blockIdx.x;
  const int nt0  = (blockIdx.y * (blockDim.x >> 5) + wave) * JT;  // SGPR
  if (nt0 * 16 >= N) return;      // scalar guard (whole wave)

  const int arow = mt * 16 + mn;  // A row held by this lane
  const bf16_t* Arow = A + (size_t)arow * K;

  const bf16_t* Brow[JT];
  v8f acc[JT];
#pragma unroll
  for (int j = 0; j < JT; ++j) {
    const int bcol = (nt0 + j) * 16 + mn;   // B column held by this lane
    Brow[j] = Bt + (size_t)bcol * K;
    if (Cin) {
#pragma unroll
      for (int v = 0; v < 8; ++v)
        acc[j][v] =
            Cin[(size_t)(mt * 16 + half * 8 + v) * N + (nt0 + j) * 16 + mn];
    } else {
#pragma unroll
      for (int v = 0; v < 8; ++v) acc[j][v] = 0.0f;
    }
  }

  for (int k0 = 0; k0 < K; k0 += 32) {
    // A 16x32 bf16 fragment: lanes 0-15 hold K = {0..7, 16..23}, lanes 16-31
    // hold K = {8..15, 24..31}  -> two contiguous 16B loads per lane.
    v8bf a0 = *(const v8bf*)(Arow + k0 + half * 8);
    v8bf a1 = *(const v8bf*)(Arow + k0 + 16 + half * 8);
    v16bf a;
#pragma unroll
    for (int i = 0; i < 8; ++i) { a[i] = a0[i]; a[i + 8] = a1[i]; }

#pragma unroll
    for (int j = 0; j < JT; ++j) {
      // B 32x16 fragment via transposed weight: lane's column slice
      // K = {half*8 + 0..7, 16 + half*8 + 0..7}, same order as A.
      v8bf b0 = *(const v8bf*)(Brow[j] + k0 + half * 8);
      v8bf b1 = *(const v8bf*)(Brow[j] + k0 + 16 + half * 8);
      v16bf b;
#pragma unroll
      for (int i = 0; i < 8; ++i) { b[i] = b0[i]; b[i + 8] = b1[i]; }
      acc[j] = __builtin_amdgcn_wmma_f32_16x16x32_bf16(
          false, a, false, b, (short)0, acc[j], false, false);
    }
  }

#pragma unroll
  for (int j = 0; j < JT; ++j) {
#pragma unroll
    for (int v = 0; v < 8; ++v)
      D[(size_t)(mt * 16 + half * 8 + v) * N + (nt0 + j) * 16 + mn] = acc[j][v];
  }
}

// ---------------------------------------------------------------------------
// Convert f32 W[K,N] -> bf16 Wt[N,K] (transpose so GEMM B-loads are contiguous)
// ---------------------------------------------------------------------------
__global__ void convert_transpose(const float* __restrict__ W,
                                  bf16_t* __restrict__ Wt, int K, int N)
{
  int idx = blockIdx.x * blockDim.x + threadIdx.x;
  if (idx >= K * N) return;
  int n = idx / K, k = idx % K;
  Wt[idx] = (bf16_t)W[(size_t)k * N + n];
}

// ---------------------------------------------------------------------------
// h[b,l,c] = c<128 ? x[b,l,:] @ W_embed[:,c] + b_embed[c] : pos_enc[l,c-128]
// ---------------------------------------------------------------------------
__global__ void embed_kernel(const float* __restrict__ x,
                             const float* __restrict__ pos,
                             const float* __restrict__ We,
                             const float* __restrict__ be,
                             float* __restrict__ h)
{
  int idx = blockIdx.x * blockDim.x + threadIdx.x;
  if (idx >= ROWS * DMODEL) return;
  int c = idx & (DMODEL - 1);
  int row = idx >> 8;             // DMODEL == 256
  int l = row & (LL - 1);
  float v;
  if (c < DMODEL / 2)
    v = x[row * 2 + 0] * We[c] + x[row * 2 + 1] * We[DMODEL / 2 + c] + be[c];
  else
    v = pos[(size_t)l * (DMODEL / 2) + (c - DMODEL / 2)];
  h[idx] = v;
}

// ---------------------------------------------------------------------------
// RMSNorm over DMODEL, output bf16 (GEMM A operand). One 256-thread block/row.
// ---------------------------------------------------------------------------
__global__ __launch_bounds__(256) void rmsnorm_kernel(
    const float* __restrict__ h, const float* __restrict__ lnw,
    bf16_t* __restrict__ xn)
{
  __shared__ float red[256];
  int row = blockIdx.x, c = threadIdx.x;
  float v = h[(size_t)row * DMODEL + c];
  red[c] = v * v;
  __syncthreads();
  for (int s = 128; s > 0; s >>= 1) {
    if (c < s) red[c] += red[c + s];
    __syncthreads();
  }
  float r = rsqrtf(red[0] * (1.0f / DMODEL) + 1e-5f);
  xn[(size_t)row * DMODEL + c] = (bf16_t)(v * r * lnw[c]);
}

// ---------------------------------------------------------------------------
// Depthwise causal conv (K=4) + bias + SiLU. Reads hid half of hidgate.
// ---------------------------------------------------------------------------
__global__ void conv_silu_kernel(const float* __restrict__ hg,
                                 const float* __restrict__ cw,
                                 const float* __restrict__ cb,
                                 float* __restrict__ hid,
                                 bf16_t* __restrict__ hidb)
{
  int idx = blockIdx.x * blockDim.x + threadIdx.x;
  if (idx >= ROWS * DINNER) return;
  int d = idx & (DINNER - 1);
  int row = idx >> 9;             // DINNER == 512
  int l = row & (LL - 1);
  int base = row - l;             // b*LL
  float acc = cb[d];
#pragma unroll
  for (int j = 0; j < KCONV; ++j) {
    int lt = l - (KCONV - 1) + j;
    if (lt >= 0)
      acc += hg[(size_t)(base + lt) * (2 * DINNER) + d] * cw[d * KCONV + j];
  }
  float s = acc / (1.0f + __expf(-acc));    // SiLU
  hid[idx]  = s;
  hidb[idx] = (bf16_t)s;
}

// ---------------------------------------------------------------------------
// delta[b,l,d] = softplus(sp[b,l,:16] @ W_dt[:,d] + b_dt[d])
// ---------------------------------------------------------------------------
__global__ void delta_kernel(const float* __restrict__ sp,
                             const float* __restrict__ Wdt,
                             const float* __restrict__ bdt,
                             float* __restrict__ dlt)
{
  int idx = blockIdx.x * blockDim.x + threadIdx.x;
  if (idx >= ROWS * DINNER) return;
  int d = idx & (DINNER - 1);
  int row = idx >> 9;
  float acc = bdt[d];
#pragma unroll
  for (int r = 0; r < DTRANK; ++r)
    acc += sp[(size_t)row * SPDIM + r] * Wdt[r * DINNER + d];
  dlt[idx] = (acc > 20.0f) ? acc : log1pf(__expf(acc));
}

__global__ void negA_kernel(const float* __restrict__ Alog,
                            float* __restrict__ negA)
{
  int idx = blockIdx.x * blockDim.x + threadIdx.x;
  if (idx < DINNER * NSTATE) negA[idx] = -__expf(Alog[idx]);
}

// ---------------------------------------------------------------------------
// Selective scan: one thread per (b,d) channel, 16-state in registers.
// Per step: stage B/C (shared across d) through LDS, update state, output
// y = (scan + hid*D_skip) * silu(gate), written bf16 for the W_out GEMM.
// ---------------------------------------------------------------------------
__global__ __launch_bounds__(256) void scan_kernel(
    const float* __restrict__ dlt, const float* __restrict__ sp,
    const float* __restrict__ hid, const float* __restrict__ hg,
    const float* __restrict__ negA, const float* __restrict__ Dsk,
    bf16_t* __restrict__ ybf)
{
  __shared__ float sBC[2 * NSTATE];
  const int b = blockIdx.x;
  const int d = blockIdx.y * 256 + threadIdx.x;
  float hs[NSTATE], nA[NSTATE];
#pragma unroll
  for (int n = 0; n < NSTATE; ++n) {
    hs[n] = 0.0f;
    nA[n] = negA[d * NSTATE + n];
  }
  const float dskip = Dsk[d];
  for (int t = 0; t < LL; ++t) {
    const int row = b * LL + t;
    __syncthreads();
    if (threadIdx.x < 2 * NSTATE)
      sBC[threadIdx.x] = sp[(size_t)row * SPDIM + DTRANK + threadIdx.x];
    __syncthreads();
    float de = dlt[(size_t)row * DINNER + d];
    float u  = hid[(size_t)row * DINNER + d];
    float du = de * u;
    float y = 0.0f;
#pragma unroll
    for (int n = 0; n < NSTATE; ++n) {
      float da = __expf(de * nA[n]);
      hs[n] = da * hs[n] + du * sBC[n];
      y += hs[n] * sBC[NSTATE + n];
    }
    y += u * dskip;
    float g = hg[(size_t)row * (2 * DINNER) + DINNER + d];
    float yo = y * (g / (1.0f + __expf(-g)));
    ybf[(size_t)row * DINNER + d] = (bf16_t)yo;
  }
}

// ---------------------------------------------------------------------------
// Heads: out[:,0] = h@W_amp + b_amp ; out[:,1] = tanh(h@W_phase + b_phase)
// ---------------------------------------------------------------------------
__global__ __launch_bounds__(256) void head_kernel(
    const float* __restrict__ h, const float* __restrict__ Wa,
    const float* __restrict__ ba, const float* __restrict__ Wp,
    const float* __restrict__ bp, float* __restrict__ out)
{
  __shared__ float ra[256], rp[256];
  int row = blockIdx.x, c = threadIdx.x;
  float v = h[(size_t)row * DMODEL + c];
  ra[c] = v * Wa[c];
  rp[c] = v * Wp[c];
  __syncthreads();
  for (int s = 128; s > 0; s >>= 1) {
    if (c < s) { ra[c] += ra[c + s]; rp[c] += rp[c + s]; }
    __syncthreads();
  }
  if (c == 0) {
    out[row * 2 + 0] = ra[0] + ba[0];
    out[row * 2 + 1] = tanhf(rp[0] + bp[0]);
  }
}

// ---------------------------------------------------------------------------
extern "C" void kernel_launch(void* const* d_in, const int* in_sizes, int n_in,
                              void* d_out, int out_size, void* d_ws, size_t ws_size,
                              hipStream_t stream)
{
  (void)in_sizes; (void)n_in; (void)out_size; (void)ws_size;

  const float* x    = (const float*)d_in[0];
  const float* pos  = (const float*)d_in[1];
  const float* We   = (const float*)d_in[2];
  const float* be   = (const float*)d_in[3];
  const float* lnw  = (const float*)d_in[4];
  const float* Win  = (const float*)d_in[5];
  const float* cw   = (const float*)d_in[6];
  const float* cb   = (const float*)d_in[7];
  const float* Wx   = (const float*)d_in[8];
  const float* Wdt  = (const float*)d_in[9];
  const float* bdt  = (const float*)d_in[10];
  const float* Alog = (const float*)d_in[11];
  const float* Dsk  = (const float*)d_in[12];
  const float* Wout = (const float*)d_in[13];
  const float* Wa   = (const float*)d_in[14];
  const float* ba   = (const float*)d_in[15];
  const float* Wp   = (const float*)d_in[16];
  const float* bp   = (const float*)d_in[17];
  float* out = (float*)d_out;

  char* wsp = (char*)d_ws;
  size_t off = 0;
  auto alloc = [&](size_t bytes) -> char* {
    char* p = wsp + off;
    off = (off + bytes + 255) & ~(size_t)255;
    return p;
  };
  float*  h     = (float*) alloc((size_t)ROWS * DMODEL * 4);
  bf16_t* xn    = (bf16_t*)alloc((size_t)ROWS * DMODEL * 2);
  float*  hg    = (float*) alloc((size_t)ROWS * 2 * DINNER * 4);
  float*  hid   = (float*) alloc((size_t)ROWS * DINNER * 4);
  bf16_t* hidb  = (bf16_t*)alloc((size_t)ROWS * DINNER * 2);
  float*  spb   = (float*) alloc((size_t)ROWS * SPDIM * 4);
  float*  dltb  = (float*) alloc((size_t)ROWS * DINNER * 4);
  bf16_t* ybf   = (bf16_t*)alloc((size_t)ROWS * DINNER * 2);
  float*  negA  = (float*) alloc((size_t)DINNER * NSTATE * 4);
  bf16_t* WinT  = (bf16_t*)alloc((size_t)NLAYERS * 2 * DINNER * DMODEL * 2);
  bf16_t* WxT   = (bf16_t*)alloc((size_t)NLAYERS * SPDIM * DINNER * 2);
  bf16_t* WoutT = (bf16_t*)alloc((size_t)NLAYERS * DMODEL * DINNER * 2);

  // Pre-transpose + down-convert all layer weights to bf16 (N x K layout).
  for (int i = 0; i < NLAYERS; ++i) {
    int kn;
    kn = DMODEL * 2 * DINNER;
    convert_transpose<<<(kn + 255) / 256, 256, 0, stream>>>(
        Win + (size_t)i * kn, WinT + (size_t)i * kn, DMODEL, 2 * DINNER);
    kn = DINNER * SPDIM;
    convert_transpose<<<(kn + 255) / 256, 256, 0, stream>>>(
        Wx + (size_t)i * kn, WxT + (size_t)i * kn, DINNER, SPDIM);
    kn = DINNER * DMODEL;
    convert_transpose<<<(kn + 255) / 256, 256, 0, stream>>>(
        Wout + (size_t)i * kn, WoutT + (size_t)i * kn, DINNER, DMODEL);
  }

  embed_kernel<<<(ROWS * DMODEL + 255) / 256, 256, 0, stream>>>(x, pos, We, be, h);

  for (int i = 0; i < NLAYERS; ++i) {
    rmsnorm_kernel<<<ROWS, 256, 0, stream>>>(h, lnw + i * DMODEL, xn);

    {   // hid|gate = xn @ W_in : 4096 x 1024 x 256
        // 64 N-tiles; 8 waves * 4 tiles = 32 tiles/block -> grid.y = 2, full.
      dim3 g(ROWS / 16, 2);
      wmma_gemm_bf16<4><<<g, 256, 0, stream>>>(
          xn, WinT + (size_t)i * DMODEL * 2 * DINNER, nullptr, hg,
          ROWS, 2 * DINNER, DMODEL);
    }

    conv_silu_kernel<<<(ROWS * DINNER + 255) / 256, 256, 0, stream>>>(
        hg, cw + i * DINNER * KCONV, cb + i * DINNER, hid, hidb);

    {   // sp = hid @ W_x : 4096 x 48 x 512
        // 3 N-tiles; one 32-thread (single-wave) block covers all 3. No tail.
      dim3 g(ROWS / 16, 1);
      wmma_gemm_bf16<3><<<g, 32, 0, stream>>>(
          hidb, WxT + (size_t)i * SPDIM * DINNER, nullptr, spb,
          ROWS, SPDIM, DINNER);
    }

    delta_kernel<<<(ROWS * DINNER + 255) / 256, 256, 0, stream>>>(
        spb, Wdt + i * DTRANK * DINNER, bdt + i * DINNER, dltb);

    negA_kernel<<<(DINNER * NSTATE + 255) / 256, 256, 0, stream>>>(
        Alog + (size_t)i * DINNER * NSTATE, negA);

    {   // sequential selective scan + skip + gating -> y (bf16)
      dim3 g(BB, DINNER / 256);
      scan_kernel<<<g, 256, 0, stream>>>(dltb, spb, hid, hg, negA,
                                         Dsk + i * DINNER, ybf);
    }

    {   // h = y @ W_out + h (residual folded in as beta=1, in place)
        // 16 N-tiles; waves 0-3 take 4 tiles each, waves 4-7 exit early.
      dim3 g(ROWS / 16, 1);
      wmma_gemm_bf16<4><<<g, 256, 0, stream>>>(
          ybf, WoutT + (size_t)i * DMODEL * DINNER, h, h,
          ROWS, DMODEL, DINNER);
    }
  }

  head_kernel<<<ROWS, 256, 0, stream>>>(h, Wa, ba, Wp, bp, out);
}
